// RNN_472446403007
// MI455X (gfx1250) — compile-verified, matrix operand order checked by reading, not run on hardware
//
#include <hip/hip_runtime.h>
#include <hip/hip_bf16.h>
#include <stdint.h>

// ---------------------------------------------------------------------------
// Leaky-ReLU rate RNN:  B=64, T=1000, I=128, R=512, O=64
//   x_proj = inp @ W_in^T + b_in
//   scan:   h = x_t + r @ W_rec^T + b_rec + 0.1*e_t ; r = 0.9 r + 0.1 relu(h)
//   out    = fr @ W_out^T + b_out
// d_out = [ out (64*1000*64 f32) | fr (64*1000*512 f32) ]
// ---------------------------------------------------------------------------

typedef _Float16 v16h __attribute__((ext_vector_type(16)));
typedef _Float16 v8h  __attribute__((ext_vector_type(8)));
typedef float    v8f  __attribute__((ext_vector_type(8)));
typedef unsigned int v4u __attribute__((ext_vector_type(4)));
typedef int v4i __attribute__((ext_vector_type(4)));
typedef int v8i __attribute__((ext_vector_type(8)));

union ABFrag { v16h v; v8h h[2]; };

#define RNN_B 64
#define RNN_T 1000
#define RNN_I 128
#define RNN_R 512
#define RNN_O 64

#define WA 520                            // padded r row stride (halves): 512 + 8
#define SMEM_REC (64 * WA * 2)            // 66,560 B : r state only
#define SMEM_OUT ((64 * WA + 128 * WA) * 2) // 199,680 B

__global__ void rnn_init_counter(unsigned* counter) { *counter = 0u; }

// TDM DMA of the 64KB f16 r-buffer (64x512 contiguous in global) into LDS,
// inserting 16B padding after every 1024B row to match the bank-conflict-
// padded A_lds layout (row stride WA halves).
__device__ __forceinline__ void tdm_reload_r(unsigned lds_off, const _Float16* rbuf) {
    unsigned long long ga = (unsigned long long)(uintptr_t)rbuf;
    v4u g0;
    g0[0] = 1u;                                               // count=1 valid descriptor
    g0[1] = lds_off;                                          // lds_addr
    g0[2] = (unsigned)ga;                                     // global_addr[31:0]
    g0[3] = (unsigned)((ga >> 32) & 0x1FFFFFFu) | (2u << 30); // global_addr[56:32], type=2
    v8i g1;
    // data_size=3 (8B), pad_enable=1, pad_interval=7 (256 DW = 1024B), pad_amount=3 (4 DW = 16B)
    g1[0] = (3 << 16) | (1 << 20) | (7 << 22) | (3 << 25);
    g1[1] = (int)((8192u & 0xFFFFu) << 16);                   // tensor_dim0[15:0] = 8192 elems
    g1[2] = (int)((8192u >> 16) | (1u << 16));                // tensor_dim0[31:16], tensor_dim1=1
    g1[3] = (int)(8192u << 16);                               // tile_dim0 = 8192 (x 8B = 64KB)
    g1[4] = 1;                                                // tile_dim1=1, tile_dim2=0
    g1[5] = 8192;                                             // tensor_dim0_stride[31:0]
    g1[6] = 0;
    g1[7] = 0;
    v4i g2 = {0, 0, 0, 0};
    v4i g3 = {0, 0, 0, 0};
#if defined(__clang_major__) && (__clang_major__ >= 23)
    v8i gz = {0, 0, 0, 0, 0, 0, 0, 0};
    __builtin_amdgcn_tensor_load_to_lds(g0, g1, g2, g3, gz, 0);
#else
    __builtin_amdgcn_tensor_load_to_lds(g0, g1, g2, g3, 0);
#endif
}

// ---------------------------------------------------------------------------
// Persistent recurrent kernel: 16 WGs, each owns a 32-column slice of
// [W_rec | W_in] held ENTIRELY IN VGPRS (loop-invariant B fragments), with the
// full r state (64x512 f16, padded) in LDS. Per step: batch-load A fragments
// (ds_load_b128 clauses), fire back-to-back v_wmma, integrate, store, global
// barrier, TDM tensor_load_to_lds reload.
// ---------------------------------------------------------------------------
__global__ void __launch_bounds__(256)
rnn_recurrent(const float* __restrict__ inp, const float* __restrict__ W_in,
              const float* __restrict__ b_in, const float* __restrict__ W_rec,
              const float* __restrict__ b_rec, const float* __restrict__ noise,
              float* __restrict__ fr, _Float16* __restrict__ rbuf,
              unsigned* __restrict__ counter) {
    extern __shared__ char smem[];
    _Float16* A_lds = (_Float16*)smem;     // [64][WA] full r state, f16

    const int tid   = threadIdx.x;
    const int wg    = blockIdx.x;          // 0..15 -> column slice [wg*32, wg*32+32)
    const int lane  = tid & 31;
    const int wave  = tid >> 5;            // 0..7
    const int mtile = wave & 3;            // 4 tiles over B=64
    const int ntile = wave >> 2;           // 2 tiles over slice of 32
    const int jl    = lane & 15;
    const int hs    = lane >> 4;           // half-select (lanes 16-31)

    // r0 = 0
    for (int idx = tid; idx < 64 * WA; idx += 256) A_lds[idx] = (_Float16)0.0f;

    const int j    = wg * 32 + ntile * 16 + jl;   // global output column
    const int arow = mtile * 16 + jl;             // A-fragment row for this lane
    const int acol = hs * 8;                      // A-fragment K sub-offset

    // ---- persistent (VGPR-resident) B fragments: W_rec row + W_in row ----
    ABFrag bw[16];                                // 16 x 8 VGPRs = 128 VGPRs
#pragma unroll
    for (int ks = 0; ks < 16; ++ks) {
        const float* wp = W_rec + (size_t)j * RNN_R + ks * 32 + acol;
#pragma unroll
        for (int q = 0; q < 8; ++q) {
            bw[ks].v[q]     = (_Float16)wp[q];
            bw[ks].v[8 + q] = (_Float16)wp[16 + q];
        }
    }
    ABFrag bi[4];                                 // 32 VGPRs
#pragma unroll
    for (int ks = 0; ks < 4; ++ks) {
        const float* wp = W_in + (size_t)j * RNN_I + ks * 32 + acol;
#pragma unroll
        for (int q = 0; q < 8; ++q) {
            bi[ks].v[q]     = (_Float16)wp[q];
            bi[ks].v[8 + q] = (_Float16)wp[16 + q];
        }
    }
    const float bias = b_rec[j] + b_in[j];

    const unsigned a_lds_off = (unsigned)(uintptr_t)A_lds;  // LDS byte offset

    v8f rfrag = {};                               // persistent r tile (C layout)
    __syncthreads();

    for (int t = 0; t < RNN_T; ++t) {
        // Issue long-latency global loads first (hide under recurrent wmmas).
        const float nz = noise[(size_t)t * RNN_R + j];
        ABFrag ai[4];
        const float* ip = inp + ((size_t)arow * RNN_T + t) * RNN_I;
#pragma unroll
        for (int ks = 0; ks < 4; ++ks) {
            const int c0 = ks * 32 + acol;
#pragma unroll
            for (int q = 0; q < 8; ++q) {
                ai[ks].v[q]     = (_Float16)ip[c0 + q];
                ai[ks].v[8 + q] = (_Float16)ip[c0 + 16 + q];
            }
        }

        v8f acc = {};
        // ---- recurrent part: r @ W_rec^T, K=512. Two chunks of 8 k-steps:
        // batch 16 ds_load_b128 then 8 back-to-back wmmas per chunk. ----
#pragma unroll
        for (int half = 0; half < 2; ++half) {
            ABFrag av[8];                         // 64 VGPRs in flight
#pragma unroll
            for (int ks = 0; ks < 8; ++ks) {
                const int c0 = (half * 8 + ks) * 32 + acol;
                av[ks].h[0] = *(const v8h*)&A_lds[arow * WA + c0];
                av[ks].h[1] = *(const v8h*)&A_lds[arow * WA + c0 + 16];
            }
#pragma unroll
            for (int ks = 0; ks < 8; ++ks) {
                acc = __builtin_amdgcn_wmma_f32_16x16x32_f16(
                    false, av[ks].v, false, bw[half * 8 + ks].v,
                    (short)0, acc, false, false);
            }
        }
        // ---- input projection: inp[:,t,:] @ W_in^T, K=128 ----
#pragma unroll
        for (int ks = 0; ks < 4; ++ks) {
            acc = __builtin_amdgcn_wmma_f32_16x16x32_f16(
                false, ai[ks].v, false, bi[ks].v, (short)0, acc, false, false);
        }
        // ---- leaky-ReLU integration (bias + noise are per-column/lane) ----
        const float add = bias + 0.1f * nz;
#pragma unroll
        for (int v = 0; v < 8; ++v) {
            float h = acc[v] + add;
            h = h > 0.0f ? h : 0.0f;
            rfrag[v] = 0.9f * rfrag[v] + 0.1f * h;
        }
        // ---- store fr (f32 -> d_out) and r slice (f16 -> exchange buffer) ----
#pragma unroll
        for (int v = 0; v < 8; ++v) {
            const int bi_row = mtile * 16 + hs * 8 + v;   // batch row (C layout)
            fr[((size_t)bi_row * RNN_T + t) * RNN_R + j] = rfrag[v];
            rbuf[bi_row * RNN_R + j] = (_Float16)rfrag[v];
        }
        // ---- device-wide barrier over the 16 co-resident WGs ----
        __syncthreads();
        if (tid == 0) {
            __threadfence();                      // publish rbuf stores
            atomicAdd(counter, 1u);
            const unsigned target = 16u * (unsigned)(t + 1);
            while (*(volatile unsigned*)counter < target) __builtin_amdgcn_s_sleep(1);
            __threadfence();                      // acquire others' stores
        }
        __syncthreads();
        // ---- TDM: DMA full 64KB r state back into padded A_lds (wave 0) ----
        if (wave == 0) {
            tdm_reload_r(a_lds_off, rbuf);
            __builtin_amdgcn_s_wait_tensorcnt(0);
        }
        __syncthreads();
    }
}

// ---------------------------------------------------------------------------
// Output projection: out[row, o] = fr[row, :] @ W_out[o, :] + b_out[o]
// row = b*T + t (64000 rows). 500 blocks x 128 rows, 8 waves each own one
// 16-row M-tile across all four 16-col N-tiles; K = 512 in 16 wmma steps.
// ---------------------------------------------------------------------------
__global__ void __launch_bounds__(256)
rnn_out_gemm(const float* __restrict__ fr, const float* __restrict__ W_out,
             const float* __restrict__ b_out, float* __restrict__ out) {
    extern __shared__ char smem[];
    _Float16* Wl = (_Float16*)smem;                   // [64][WA]  W_out f16
    _Float16* Al = (_Float16*)(smem + 64 * WA * 2);   // [128][WA] fr block f16

    const int tid  = threadIdx.x;
    const int lane = tid & 31;
    const int wave = tid >> 5;
    const size_t row0 = (size_t)blockIdx.x * 128;

    for (int idx = tid; idx < 64 * 512; idx += 256) {
        int o = idx >> 9, c = idx & 511;
        Wl[o * WA + c] = (_Float16)W_out[idx];
    }
    for (int idx = tid; idx < 128 * 512; idx += 256) {
        int rl = idx >> 9, c = idx & 511;
        Al[rl * WA + c] = (_Float16)fr[(row0 + rl) * RNN_R + c];
    }
    __syncthreads();

    const int jl = lane & 15, hs = lane >> 4;
    const int arow = wave * 16 + jl;
    v8f acc[4] = {{}, {}, {}, {}};
#pragma unroll 2
    for (int ks = 0; ks < 16; ++ks) {
        const int c0 = ks * 32 + hs * 8;
        ABFrag af;
        af.h[0] = *(const v8h*)&Al[arow * WA + c0];
        af.h[1] = *(const v8h*)&Al[arow * WA + c0 + 16];
#pragma unroll
        for (int nt = 0; nt < 4; ++nt) {
            ABFrag bf;
            const int br = nt * 16 + jl;
            bf.h[0] = *(const v8h*)&Wl[br * WA + c0];
            bf.h[1] = *(const v8h*)&Wl[br * WA + c0 + 16];
            acc[nt] = __builtin_amdgcn_wmma_f32_16x16x32_f16(false, af.v, false, bf.v,
                                                             (short)0, acc[nt], false, false);
        }
    }
#pragma unroll
    for (int nt = 0; nt < 4; ++nt) {
        const int o = nt * 16 + jl;
        const float bo = b_out[o];
#pragma unroll
        for (int v = 0; v < 8; ++v) {
            const size_t row = row0 + wave * 16 + hs * 8 + v;
            out[row * RNN_O + o] = acc[nt][v] + bo;
        }
    }
}

extern "C" void kernel_launch(void* const* d_in, const int* in_sizes, int n_in,
                              void* d_out, int out_size, void* d_ws, size_t ws_size,
                              hipStream_t stream) {
    const float* inp   = (const float*)d_in[0];
    const float* W_in  = (const float*)d_in[1];
    const float* b_in  = (const float*)d_in[2];
    const float* W_rec = (const float*)d_in[3];
    const float* b_rec = (const float*)d_in[4];
    const float* W_out = (const float*)d_in[5];
    const float* b_out = (const float*)d_in[6];
    const float* noise = (const float*)d_in[7];

    float* out = (float*)d_out;                                  // [B,T,O]
    float* fr  = out + (size_t)RNN_B * RNN_T * RNN_O;            // [B,T,R]

    _Float16* rbuf    = (_Float16*)d_ws;                         // 64KB r exchange
    unsigned* counter = (unsigned*)((char*)d_ws + 65536);        // barrier counter

    rnn_init_counter<<<1, 1, 0, stream>>>(counter);
    rnn_recurrent<<<16, 256, SMEM_REC, stream>>>(inp, W_in, b_in, W_rec, b_rec,
                                                 noise, fr, rbuf, counter);
    rnn_out_gemm<<<(RNN_B * RNN_T) / 128, 256, SMEM_OUT, stream>>>(fr, W_out, b_out, out);
}
